// ConditionalRandomField_82240033784449
// MI455X (gfx1250) — matrix operand order repaired and checked by reading, compiled, not run on hardware
//
#include <hip/hip_runtime.h>

#define B_TOT 512
#define S_LEN 512
#define T_TAG 128
#define NEG_INF (-10000.0f)
#define LOG2E 1.4426950408889634f
#define LN2   0.6931471805599453f

typedef __attribute__((ext_vector_type(16))) _Float16 v16h;
typedef __attribute__((ext_vector_type(8)))  float    v8f;

// ---------------------------------------------------------------------------
// Kernel 0: E[i,j] = exp(transition[i,j]) pre-swizzled into the WMMA B-fragment
// layout as used here (16-bit B, 32x16 K x N tile, wave32; contiguous k, same as the A packing):
//   lane L holds column N = nt*16 + (L&15), K = kt*32 + (L&16) + h, h = 0..15
//   epack[((kt*8 + nt)*32 + L)*16 + h]
// ---------------------------------------------------------------------------
__global__ void crf_build_epack(const float* __restrict__ trans,
                                _Float16* __restrict__ epack) {
    int idx = blockIdx.x * blockDim.x + threadIdx.x;   // 0 .. 16383
    int h  = idx & 15;
    int L  = (idx >> 4) & 31;
    int nt = (idx >> 9) & 7;
    int kt = idx >> 12;
    int n = nt * 16 + (L & 15);
    int k = kt * 32 + (L & 16) + h;
    epack[idx] = (_Float16)__builtin_amdgcn_exp2f(trans[n * T_TAG + k] * LOG2E);
}

// ---------------------------------------------------------------------------
// Kernel 1: CRF forward recursion. One block = 16 batch rows, 8 waves.
// Wave w owns output columns [16w, 16w+16). B-fragments of E live in VGPRs
// for the whole S loop; A-fragments (exp2(alpha*log2e - rowmax*log2e)) are
// rebuilt in LDS each step in fragment-friendly order. Alpha footprint of
// each wave's C-tile is mirrored in 8 VGPRs for a branchless masked update.
// ---------------------------------------------------------------------------
__global__ __launch_bounds__(256) void crf_forward(
    const float* __restrict__ x, const float* __restrict__ mask,
    const float* __restrict__ trans, const _Float16* __restrict__ epack,
    float* __restrict__ fwd) {

    __shared__ float alpha[16 * 128];
    __shared__ __attribute__((aligned(16))) float rowmax[16];
    __shared__ __attribute__((aligned(16))) float mrow[16];
    __shared__ __attribute__((aligned(32))) _Float16 pfrag[4 * 32 * 16];

    const int tid   = threadIdx.x;
    const int lane  = tid & 31;
    const int w     = tid >> 5;          // wave id == N-tile id
    const int b0    = blockIdx.x * 16;
    const int hf    = lane >> 4;         // 0: lanes 0-15, 1: lanes 16-31
    const int lan16 = lane & 15;
    const int mred  = 2 * w + hf;        // row this lane-group reduces
    const int mbase = hf * 8;            // C-tile row offset for this lane
    const int n     = w * 16 + lan16;    // column owned by this lane

    // init alpha: 0 at START_ID(=0), NEG elsewhere
    for (int e = tid; e < 16 * 128; e += 256)
        alpha[e] = ((e & 127) == 0) ? 0.0f : NEG_INF;

    // register mirror of this wave's alpha footprint (C-tile layout)
    float areg[8];
#pragma unroll
    for (int r = 0; r < 8; ++r) areg[r] = (n == 0) ? 0.0f : NEG_INF;

    // load constant B fragments of E into VGPRs (kept live across S loop)
    v16h bfrag[4];
#pragma unroll
    for (int kt = 0; kt < 4; ++kt)
        bfrag[kt] = *(const v16h*)(epack + (((kt * 8 + w) * 32 + lane) << 4));

    for (int s = 0; s < S_LEN; ++s) {
        __syncthreads();                       // alpha/rowmax/mrow stable

        // ---- issue this step's x loads early (latency hidden by phases 1-3)
        const float* xp = x + ((size_t)(b0 + mbase) * S_LEN + s) * T_TAG + n;
        float xv[8];
#pragma unroll
        for (int r = 0; r < 8; ++r)
            xv[r] = xp[(size_t)r * S_LEN * T_TAG];
        if (tid < 16) mrow[tid] = mask[(size_t)(b0 + tid) * S_LEN + s];

        // ---- per-row max of alpha (wave w: rows 2w, 2w+1) ----
        float mx = NEG_INF;
#pragma unroll
        for (int q = 0; q < 8; ++q)
            mx = fmaxf(mx, alpha[mred * 128 + lan16 + q * 16]);
#pragma unroll
        for (int o = 8; o >= 1; o >>= 1)
            mx = fmaxf(mx, __shfl_xor(mx, o, 16));
        if (lan16 == 0) rowmax[mred] = mx;
        __syncthreads();

        // ---- P = exp2(alpha*log2e - rowmax*log2e) into A-fragment layout ----
        {
            const float nmk = -rowmax[tid >> 4] * LOG2E;   // row m = tid>>4
#pragma unroll
            for (int u = 0; u < 8; ++u) {
                int e = tid * 8 + u;               // e = m*128 + k
                int k = e & 127;
                float p = __builtin_amdgcn_exp2f(fmaf(alpha[e], LOG2E, nmk));
                int kt = k >> 5, r = k & 31;
                int hh = r & 15;                       // half index in lane
                int LL = (e >> 7) + (r & 16);          // lane index
                pfrag[((kt * 32 + LL) << 4) + hh] = (_Float16)p;
            }
        }
        __syncthreads();

        // ---- C = P @ E'  (16x128 @ 128x16 per wave), f32 accumulate ----
        v8f c = {};
#pragma unroll
        for (int kt = 0; kt < 4; ++kt) {
            v16h a = *(const v16h*)(pfrag + ((kt * 32 + lane) << 4));
            c = __builtin_amdgcn_wmma_f32_16x16x32_f16(
                    false, a, false, bfrag[kt], (short)0, c, false, false);
        }

        // ---- branchless masked alpha update (cndmask, no exec branches) ----
        float rm[8], mk[8];
        *(float4*)&rm[0] = *(const float4*)&rowmax[mbase];
        *(float4*)&rm[4] = *(const float4*)&rowmax[mbase + 4];
        *(float4*)&mk[0] = *(const float4*)&mrow[mbase];
        *(float4*)&mk[4] = *(const float4*)&mrow[mbase + 4];
#pragma unroll
        for (int r = 0; r < 8; ++r) {
            float newv = xv[r] + rm[r] + LN2 * __builtin_amdgcn_logf(c[r]);
            areg[r] = (mk[r] != 0.0f) ? newv : areg[r];
            alpha[(r + mbase) * 128 + n] = areg[r];
        }
    }
    __syncthreads();

    // ---- fwd_score[b] = logsumexp_i(alpha[b,i] + transition[END_ID=1, i]) ----
    float mx = NEG_INF;
    float vals[8];
#pragma unroll
    for (int q = 0; q < 8; ++q) {
        int j = lan16 + q * 16;
        vals[q] = alpha[mred * 128 + j] + trans[1 * T_TAG + j];
        mx = fmaxf(mx, vals[q]);
    }
#pragma unroll
    for (int o = 8; o >= 1; o >>= 1)
        mx = fmaxf(mx, __shfl_xor(mx, o, 16));
    float sum = 0.0f;
#pragma unroll
    for (int q = 0; q < 8; ++q)
        sum += __builtin_amdgcn_exp2f((vals[q] - mx) * LOG2E);
#pragma unroll
    for (int o = 8; o >= 1; o >>= 1)
        sum += __shfl_xor(sum, o, 16);
    if (lan16 == 0) fwd[b0 + mred] = mx + LN2 * __builtin_amdgcn_logf(sum);
}

// ---------------------------------------------------------------------------
// Kernel 2: gold path score. One wave32 per batch row.
// ---------------------------------------------------------------------------
__global__ __launch_bounds__(256) void crf_gold(
    const float* __restrict__ x, const int* __restrict__ tags,
    const float* __restrict__ mask, const float* __restrict__ trans,
    float* __restrict__ gold) {
    int lane = threadIdx.x & 31;
    int b    = blockIdx.x * 8 + (threadIdx.x >> 5);
    const int*   tg = tags + (size_t)b * S_LEN;
    const float* mk = mask + (size_t)b * S_LEN;
    float sc = 0.0f, ms = 0.0f;
    for (int s = lane; s < S_LEN; s += 32) {
        float m = mk[s];
        ms += m;
        if (s < S_LEN - 1) {
            int tn = tg[s + 1];
            int tc = tg[s];
            sc += (x[((size_t)b * S_LEN + s) * T_TAG + tn] +
                   trans[tn * T_TAG + tc]) * m;
        }
    }
#pragma unroll
    for (int o = 16; o >= 1; o >>= 1) {
        sc += __shfl_xor(sc, o, 32);
        ms += __shfl_xor(ms, o, 32);
    }
    if (lane == 0) {
        int last = (int)(ms + 0.5f);           // lengths[b], exact in f32
        gold[b] = sc + trans[1 * T_TAG + tg[last]];
    }
}

// ---------------------------------------------------------------------------
// Kernel 3: out = mean(fwd - gold), deterministic tree reduce.
// ---------------------------------------------------------------------------
__global__ void crf_reduce(const float* __restrict__ fwd,
                           const float* __restrict__ gold,
                           float* __restrict__ out) {
    __shared__ float buf[256];
    int t = threadIdx.x;
    buf[t] = (fwd[t] - gold[t]) + (fwd[t + 256] - gold[t + 256]);
    __syncthreads();
    for (int o = 128; o >= 1; o >>= 1) {
        if (t < o) buf[t] += buf[t + o];
        __syncthreads();
    }
    if (t == 0) out[0] = buf[0] / (float)B_TOT;
}

extern "C" void kernel_launch(void* const* d_in, const int* in_sizes, int n_in,
                              void* d_out, int out_size, void* d_ws, size_t ws_size,
                              hipStream_t stream) {
    (void)in_sizes; (void)n_in; (void)out_size; (void)ws_size;
    const float* x     = (const float*)d_in[0];
    const int*   tags  = (const int*)d_in[1];
    const float* mask  = (const float*)d_in[2];
    const float* trans = (const float*)d_in[3];

    _Float16* epack = (_Float16*)d_ws;                     // 32 KB
    float*    fwd   = (float*)((char*)d_ws + 32768);       // 512 f32
    float*    gold  = fwd + B_TOT;                         // 512 f32

    crf_build_epack<<<64, 256, 0, stream>>>(trans, epack);
    crf_forward   <<<32, 256, 0, stream>>>(x, mask, trans, epack, fwd);
    crf_gold      <<<64, 256, 0, stream>>>(x, tags, mask, trans, gold);
    crf_reduce    <<<1, 256, 0, stream>>>(fwd, gold, (float*)d_out);
}
